// up_deeplabv3Head_43284680409314
// MI455X (gfx1250) — matrix operand order repaired and checked by reading, not compiled, not run on hardware
//
#include <hip/hip_runtime.h>
#include <hip/hip_bf16.h>

// ---------------------------------------------------------------------------
// CDNA5 (gfx1250) implementation: all convolutions lowered to implicit-GEMM
// using v_wmma_f32_16x16x32_bf16 (wave32), software-pipelined with
// double-buffered LDS so global-load latency overlaps the matrix pipe.
// f32 inputs are converted to bf16 while staging; accumulation stays in f32.
// ---------------------------------------------------------------------------

typedef __bf16 bf16_t;
typedef __bf16 v16bf __attribute__((ext_vector_type(16)));
typedef __bf16 v8bf  __attribute__((ext_vector_type(8)));
typedef __bf16 v2bf  __attribute__((ext_vector_type(2)));
typedef float  v8f   __attribute__((ext_vector_type(8)));

#define BM 128
#define BN 128
#define BKK 32

// O[MO, P] = relu( scale * ( W[MO, C*KT2] x im2col(X)[C*KT2, P] ) + bias )
// KT2 = 9 -> 3x3 conv with dilation `dil`, padding `dil` (zero pad)
// KT2 = 1 -> 1x1 conv
template <int KT2>
__global__ __launch_bounds__(256) void conv_gemm_wmma(
    const float* __restrict__ Wmat,   // [MO][C*KT2]
    const float* __restrict__ X,      // [C][H*W]
    const float* __restrict__ scale,  // [MO] or nullptr (=> 1.0)
    const float* __restrict__ bias,   // [MO]
    float* __restrict__ Out,          // [MO][H*W]
    int MO, int C, int H, int Wd, int dil, int relu) {
  __shared__ bf16_t As[2][BM][BKK];   // weights tile (row-major, M x K)
  __shared__ bf16_t Bs[2][BN][BKK];   // im2col tile  (transposed: pixel x K)

  const int tid  = threadIdx.x;
  const int lane = tid & 31;
  const int wv   = tid >> 5;     // 0..7
  const int wm   = wv >> 2;      // 0..1  (M direction, 64 rows each)
  const int wn   = wv & 3;       // 0..3  (N direction, 32 cols each)
  const int khi  = lane >> 4;    // lane half selects K sub-range
  const int l16  = lane & 15;

  const int P    = H * Wd;
  const int Ktot = C * KT2;
  const int m0   = blockIdx.y * BM;
  const int n0   = blockIdx.x * BN;

  // staging thread mapping (fixed per thread):
  // each thread gathers 2 adjacent K rows x 8 pixels -> packed b32 LDS stores
  const int bkk2 = (tid >> 4) << 1;   // even K row 0,2,..,30
  const int bp0  = (tid & 15) << 3;   // 8 consecutive pixels

  v8f acc[4][2];
#pragma unroll
  for (int i = 0; i < 4; ++i)
#pragma unroll
    for (int j = 0; j < 2; ++j)
#pragma unroll
      for (int e = 0; e < 8; ++e) acc[i][j][e] = 0.0f;

  // staging registers (next tile in flight while WMMA consumes current tile)
  float4 va[4];
  float  vb[2][8];

  // issue global loads for the K-chunk starting at k0 (no LDS traffic here)
  auto issue_loads = [&](int k0) {
#pragma unroll
    for (int ch = 0; ch < 4; ++ch) {
      int f   = tid + ch * 256;       // float4 id within 128x32 tile
      int row = f >> 3;
      int c4  = (f & 7) << 2;
      int gm  = m0 + row;
      va[ch] = make_float4(0.f, 0.f, 0.f, 0.f);
      if (gm < MO) {
        const float* src = Wmat + (size_t)gm * Ktot + k0 + c4;
        va[ch] = *(const float4*)src;
        if (ch == 0 && k0 + BKK < Ktot)
          __builtin_prefetch(src + BKK, 0, 1);   // global_prefetch_b8
      }
    }
#pragma unroll
    for (int kr = 0; kr < 2; ++kr) {
      int k = k0 + bkk2 + kr;
      int c = k / KT2;
      int t = k - c * KT2;
      int dy = (KT2 == 9) ? (t / 3 - 1) * dil : 0;
      int dx = (KT2 == 9) ? (t % 3 - 1) * dil : 0;
      const float* Xc = X + (size_t)c * P;
#pragma unroll
      for (int j = 0; j < 8; ++j) {
        int p = n0 + bp0 + j;
        int y = p / Wd + dy;
        int x = p - (p / Wd) * Wd + dx;
        vb[kr][j] = (y >= 0 && y < H && x >= 0 && x < Wd) ? Xc[y * Wd + x] : 0.f;
      }
    }
  };

  // convert staged registers to bf16 and commit to LDS buffer `buf`
  auto commit = [&](int buf) {
#pragma unroll
    for (int ch = 0; ch < 4; ++ch) {
      int f   = tid + ch * 256;
      int row = f >> 3;
      int c4  = (f & 7) << 2;
      As[buf][row][c4 + 0] = (bf16_t)va[ch].x;
      As[buf][row][c4 + 1] = (bf16_t)va[ch].y;
      As[buf][row][c4 + 2] = (bf16_t)va[ch].z;
      As[buf][row][c4 + 3] = (bf16_t)va[ch].w;
    }
#pragma unroll
    for (int j = 0; j < 8; ++j) {
      v2bf t;
      t[0] = (bf16_t)vb[0][j];   // K = bkk2
      t[1] = (bf16_t)vb[1][j];   // K = bkk2+1 (contiguous in [pixel][K])
      *(v2bf*)&Bs[buf][bp0 + j][bkk2] = t;   // one packed ds_store_b32
    }
  };

  // fragment loads (documented 16-bit A/B VGPR layouts) + 8 WMMAs
  auto compute = [&](int buf) {
    v16bf afrag[4];
#pragma unroll
    for (int fm = 0; fm < 4; ++fm) {
      int ml = wm * 64 + fm * 16 + l16;
      // lanes 0-15: K 0..7 then 16..23 ; lanes 16-31: K 8..15 then 24..31
      v8bf lo = *(const v8bf*)&As[buf][ml][8 * khi];
      v8bf hi = *(const v8bf*)&As[buf][ml][16 + 8 * khi];
#pragma unroll
      for (int e = 0; e < 8; ++e) { afrag[fm][e] = lo[e]; afrag[fm][e + 8] = hi[e]; }
    }
    v16bf bfrag[2];
#pragma unroll
    for (int fn = 0; fn < 2; ++fn) {
      int nl = wn * 32 + fn * 16 + l16;
      // lanes 0-15: K 0..7 then 16..23 ; lanes 16-31: K 8..15 then 24..31 (column = lane&15)
      v8bf blo = *(const v8bf*)&Bs[buf][nl][8 * khi];
      v8bf bhi = *(const v8bf*)&Bs[buf][nl][16 + 8 * khi];
#pragma unroll
      for (int e = 0; e < 8; ++e) { bfrag[fn][e] = blo[e]; bfrag[fn][e + 8] = bhi[e]; }
    }
#pragma unroll
    for (int fm = 0; fm < 4; ++fm)
#pragma unroll
      for (int fn = 0; fn < 2; ++fn)
        acc[fm][fn] = __builtin_amdgcn_wmma_f32_16x16x32_bf16(
            false, afrag[fm], false, bfrag[fn], (short)0, acc[fm][fn], false, false);
  };

  // ---- software pipeline: prologue ----
  const int nIter = Ktot / BKK;   // all K in this network are multiples of 32
  issue_loads(0);
  commit(0);
  __syncthreads();

  for (int i = 0; i < nIter; ++i) {
    int buf = i & 1;
    if (i + 1 < nIter) issue_loads((i + 1) * BKK);  // loads in flight...
    compute(buf);                                   // ...while WMMA runs
    if (i + 1 < nIter) commit(buf ^ 1);             // wait+cvt+store next buf
    __syncthreads();
  }

  // ---- fused scale/bias/relu epilogue; D: VGPR r -> M = r + 8*khi, N = l16
#pragma unroll
  for (int fm = 0; fm < 4; ++fm) {
#pragma unroll
    for (int fn = 0; fn < 2; ++fn) {
      int nb = n0 + wn * 32 + fn * 16 + l16;
#pragma unroll
      for (int r = 0; r < 8; ++r) {
        int m = m0 + wm * 64 + fm * 16 + r + 8 * khi;
        if (m < MO) {
          float s = scale ? scale[m] : 1.0f;
          float v = acc[fm][fn][r] * s + bias[m];
          if (relu) v = fmaxf(v, 0.0f);
          Out[(size_t)m * P + nb] = v;
        }
      }
    }
  }
}

// bilinear resize, align_corners=True, NCHW (n=1)
__global__ void upsample_ac(const float* __restrict__ in, float* __restrict__ out,
                            int C, int IH, int IW, int OH, int OW) {
  int idx = blockIdx.x * blockDim.x + threadIdx.x;
  int OP = OH * OW;
  if (idx >= C * OP) return;
  int c = idx / OP, r = idx - c * OP, oy = r / OW, ox = r - (r / OW) * OW;
  float ry = (OH > 1) ? (float)(IH - 1) / (float)(OH - 1) : 0.f;
  float rx = (OW > 1) ? (float)(IW - 1) / (float)(OW - 1) : 0.f;
  float ty = oy * ry, tx = ox * rx;
  int y0 = (int)floorf(ty), x0 = (int)floorf(tx);
  float fy = ty - (float)y0, fx = tx - (float)x0;
  int y1 = min(y0 + 1, IH - 1), x1 = min(x0 + 1, IW - 1);
  const float* b = in + (size_t)c * IH * IW;
  float v00 = b[y0 * IW + x0], v01 = b[y0 * IW + x1];
  float v10 = b[y1 * IW + x0], v11 = b[y1 * IW + x1];
  out[idx] = (1.f - fy) * ((1.f - fx) * v00 + fx * v01) +
             fy * ((1.f - fx) * v10 + fx * v11);
}

// per-pixel energy over 3x3 (dilation-2, zero-pad) window + softmax -> att[9][P]
__global__ void att_kernel(const float* __restrict__ c1r, const float* __restrict__ c2r,
                           float* __restrict__ att, int H, int W) {
  int p = blockIdx.x * blockDim.x + threadIdx.x;
  int P = H * W;
  if (p >= P) return;
  int y = p / W, x = p - (p / W) * W;
  float e[9];
  int off[9];
  bool ok[9];
#pragma unroll
  for (int k = 0; k < 9; ++k) {
    int yy = y + (k / 3 - 1) * 2, xx = x + (k % 3 - 1) * 2;
    ok[k] = (yy >= 0 && yy < H && xx >= 0 && xx < W);
    off[k] = yy * W + xx;
    e[k] = 0.f;   // zero-padded unfold -> energy 0 for OOB taps
  }
  for (int c = 0; c < 64; ++c) {
    float a = c1r[c * P + p];
    const float* b = c2r + c * P;
#pragma unroll
    for (int k = 0; k < 9; ++k)
      if (ok[k]) e[k] += a * b[off[k]];
  }
  float m = e[0];
#pragma unroll
  for (int k = 1; k < 9; ++k) m = fmaxf(m, e[k]);
  float s = 0.f;
#pragma unroll
  for (int k = 0; k < 9; ++k) { e[k] = __expf(e[k] - m); s += e[k]; }
  float inv = 1.f / s;
#pragma unroll
  for (int k = 0; k < 9; ++k) att[k * P + p] = e[k] * inv;
}

// global average pool over P pixels, one block per channel
__global__ void gap_kernel(const float* __restrict__ x, float* __restrict__ g, int P) {
  __shared__ float red[256];
  int c = blockIdx.x;
  float s = 0.f;
  for (int i = threadIdx.x; i < P; i += 256) s += x[(size_t)c * P + i];
  red[threadIdx.x] = s;
  __syncthreads();
  for (int st = 128; st > 0; st >>= 1) {
    if (threadIdx.x < st) red[threadIdx.x] += red[threadIdx.x + st];
    __syncthreads();
  }
  if (threadIdx.x == 0) g[c] = red[0] / (float)P;
}

// f4 = relu(s * (a4_w @ gap) + b), broadcast over P pixels; one block per out channel
__global__ void f4_kernel(const float* __restrict__ w, const float* __restrict__ s,
                          const float* __restrict__ b, const float* __restrict__ g,
                          float* __restrict__ out, int C, int P) {
  __shared__ float red[256];
  int o = blockIdx.x;
  float d = 0.f;
  for (int c = threadIdx.x; c < C; c += 256) d += w[(size_t)o * C + c] * g[c];
  red[threadIdx.x] = d;
  __syncthreads();
  for (int st = 128; st > 0; st >>= 1) {
    if (threadIdx.x < st) red[threadIdx.x] += red[threadIdx.x + st];
    __syncthreads();
  }
  __shared__ float vsh;
  if (threadIdx.x == 0) vsh = fmaxf(s[o] * red[0] + b[o], 0.f);
  __syncthreads();
  float v = vsh;
  for (int p = threadIdx.x; p < P; p += 256) out[(size_t)o * P + p] = v;
}

// agg[c][p] = sum_k att[k][p] * up[c][neighbor_k(p)]  (3x3, dilation 2, zero pad)
__global__ void agg_kernel(const float* __restrict__ att, const float* __restrict__ up,
                           float* __restrict__ agg, int H, int W, int C) {
  int idx = blockIdx.x * blockDim.x + threadIdx.x;
  int P = H * W;
  if (idx >= C * P) return;
  int c = idx / P, p = idx - c * P;
  int y = p / W, x = p - (p / W) * W;
  const float* u = up + (size_t)c * P;
  float s = 0.f;
#pragma unroll
  for (int k = 0; k < 9; ++k) {
    int yy = y + (k / 3 - 1) * 2, xx = x + (k % 3 - 1) * 2;
    if (yy >= 0 && yy < H && xx >= 0 && xx < W) s += att[k * P + p] * u[yy * W + xx];
  }
  agg[idx] = s;
}

extern "C" void kernel_launch(void* const* d_in, const int* in_sizes, int n_in,
                              void* d_out, int out_size, void* d_ws, size_t ws_size,
                              hipStream_t stream) {
  (void)in_sizes; (void)n_in; (void)out_size; (void)ws_size;
  const float* c1    = (const float*)d_in[0];   // [256,64,64]
  const float* c2    = (const float*)d_in[1];   // [512,32,32]
  const float* x     = (const float*)d_in[2];   // [4096,32,32]
  const float* rf1_w = (const float*)d_in[3];
  const float* rf1_s = (const float*)d_in[4];
  const float* rf1_b = (const float*)d_in[5];
  const float* rf2_w = (const float*)d_in[6];
  const float* rf2_s = (const float*)d_in[7];
  const float* rf2_b = (const float*)d_in[8];
  const float* r2_w  = (const float*)d_in[9];
  const float* r2_s  = (const float*)d_in[10];
  const float* r2_b  = (const float*)d_in[11];
  const float* a0_w  = (const float*)d_in[12];
  const float* a0_s  = (const float*)d_in[13];
  const float* a0_b  = (const float*)d_in[14];
  const float* a1_w  = (const float*)d_in[15];
  const float* a1_s  = (const float*)d_in[16];
  const float* a1_b  = (const float*)d_in[17];
  const float* a2_w  = (const float*)d_in[18];
  const float* a2_s  = (const float*)d_in[19];
  const float* a2_b  = (const float*)d_in[20];
  const float* a3_w  = (const float*)d_in[21];
  const float* a3_s  = (const float*)d_in[22];
  const float* a3_b  = (const float*)d_in[23];
  const float* a4_w  = (const float*)d_in[24];
  const float* a4_s  = (const float*)d_in[25];
  const float* a4_b  = (const float*)d_in[26];
  const float* ap_w  = (const float*)d_in[27];
  const float* ap_s  = (const float*)d_in[28];
  const float* ap_b  = (const float*)d_in[29];
  const float* c6_w  = (const float*)d_in[30];
  const float* c6_b  = (const float*)d_in[31];

  float* ws = (float*)d_ws;
  size_t off = 0;
  float* c2up = ws + off; off += (size_t)512 * 4096;   // upsampled c2
  float* c2r  = ws + off; off += (size_t)64 * 4096;    // refine2 output
  float* t1   = ws + off; off += (size_t)64 * 4096;    // refine stage 1
  float* c1r  = ws + off; off += (size_t)64 * 4096;    // refine output
  float* attb = ws + off; off += (size_t)9 * 4096;     // attention [9][4096]
  float* cat  = ws + off; off += (size_t)2560 * 1024;  // ASPP concat
  float* g    = ws + off; off += (size_t)4096;         // GAP
  float* outb = ws + off; off += (size_t)512 * 1024;   // ASPP projection
  float* upo  = ws + off; off += (size_t)512 * 4096;   // upsampled ASPP out
  float* agg  = ws + off; off += (size_t)512 * 4096;   // aggregated features

  // --- c2 branch: bilinear 32->64, then 1x1 conv 512->64 + bn_relu
  upsample_ac<<<(512 * 4096 + 255) / 256, 256, 0, stream>>>(c2, c2up, 512, 32, 32, 64, 64);
  conv_gemm_wmma<1><<<dim3(4096 / BN, 1), 256, 0, stream>>>(
      r2_w, c2up, r2_s, r2_b, c2r, 64, 512, 64, 64, 1, 1);

  // --- c1 branch: two 3x3 dilation-2 convs
  conv_gemm_wmma<9><<<dim3(4096 / BN, 1), 256, 0, stream>>>(
      rf1_w, c1, rf1_s, rf1_b, t1, 64, 256, 64, 64, 2, 1);
  conv_gemm_wmma<9><<<dim3(4096 / BN, 1), 256, 0, stream>>>(
      rf2_w, t1, rf2_s, rf2_b, c1r, 64, 64, 64, 64, 2, 1);

  // --- local window attention weights
  att_kernel<<<(4096 + 255) / 256, 256, 0, stream>>>(c1r, c2r, attb, 64, 64);

  // --- ASPP on x (4096 -> 512 each), written into the concat buffer
  conv_gemm_wmma<1><<<dim3(1024 / BN, 512 / BM), 256, 0, stream>>>(
      a0_w, x, a0_s, a0_b, cat + (size_t)0 * 512 * 1024, 512, 4096, 32, 32, 1, 1);
  conv_gemm_wmma<9><<<dim3(1024 / BN, 512 / BM), 256, 0, stream>>>(
      a1_w, x, a1_s, a1_b, cat + (size_t)1 * 512 * 1024, 512, 4096, 32, 32, 12, 1);
  conv_gemm_wmma<9><<<dim3(1024 / BN, 512 / BM), 256, 0, stream>>>(
      a2_w, x, a2_s, a2_b, cat + (size_t)2 * 512 * 1024, 512, 4096, 32, 32, 24, 1);
  conv_gemm_wmma<9><<<dim3(1024 / BN, 512 / BM), 256, 0, stream>>>(
      a3_w, x, a3_s, a3_b, cat + (size_t)3 * 512 * 1024, 512, 4096, 32, 32, 36, 1);
  gap_kernel<<<4096, 256, 0, stream>>>(x, g, 1024);
  f4_kernel<<<512, 256, 0, stream>>>(a4_w, a4_s, a4_b, g,
                                     cat + (size_t)4 * 512 * 1024, 4096, 1024);

  // --- ASPP projection 2560 -> 512
  conv_gemm_wmma<1><<<dim3(1024 / BN, 512 / BM), 256, 0, stream>>>(
      ap_w, cat, ap_s, ap_b, outb, 512, 2560, 32, 32, 1, 1);

  // --- upsample 32->64, attention aggregation, final 1x1 conv 512->19 (+bias)
  upsample_ac<<<(512 * 4096 + 255) / 256, 256, 0, stream>>>(outb, upo, 512, 32, 32, 64, 64);
  agg_kernel<<<(512 * 4096 + 255) / 256, 256, 0, stream>>>(attb, upo, agg, 64, 64, 512);
  conv_gemm_wmma<1><<<dim3(4096 / BN, 1), 256, 0, stream>>>(
      c6_w, agg, nullptr, c6_b, (float*)d_out, 19, 512, 64, 64, 1, 0);
}
